// Attention_34866544508962
// MI455X (gfx1250) — matrix operand order repaired and checked by reading, not compiled, not run on hardware
//
#include <hip/hip_runtime.h>

// ---------------------------------------------------------------------------
// Types / helpers
// ---------------------------------------------------------------------------
typedef __attribute__((ext_vector_type(16))) __bf16 v16bf;
typedef __attribute__((ext_vector_type(8)))  float  v8f;

union FragBF {                 // one 16x32 bf16 A-frag or 32x16 bf16 B-frag
    uint4        q[2];
    unsigned int u[8];
    v16bf        v;
};

__device__ __forceinline__ unsigned short f2bf(float f) {
    unsigned int u = __float_as_uint(f);
    u += 0x7FFFu + ((u >> 16) & 1u);          // round-to-nearest-even
    return (unsigned short)(u >> 16);
}
__device__ __forceinline__ unsigned int pack2bf(float lo, float hi) {
    return (unsigned int)f2bf(lo) | ((unsigned int)f2bf(hi) << 16);
}
__device__ __forceinline__ v8f zero8() {
    v8f z = {0.f,0.f,0.f,0.f,0.f,0.f,0.f,0.f};
    return z;
}
__device__ __forceinline__ v8f wmma_bf16(v16bf a, v16bf b, v8f c) {
    return __builtin_amdgcn_wmma_f32_16x16x32_bf16(
        /*neg_a=*/false, a, /*neg_b=*/false, b,
        /*c_mod=*/(short)0, c, /*reuse_a=*/false, /*reuse_b=*/false);
}

// ---------------------------------------------------------------------------
// Generic tiled WMMA GEMM: C[M x N](f32) = A[M x K] * B[K x N](f32)
// A is f32 (converted to bf16 in registers) or bf16, row-major, lda == K.
// Block: 256 threads = 8 waves; block tile 128(M) x 64(N); K step 32.
// B tile staged transposed into LDS as bf16 so B-fragment reads are
// two contiguous 16B runs per lane (ISA 7.12.2 16-bit B layout).
// ---------------------------------------------------------------------------
template<bool A_IS_BF16>
__global__ __launch_bounds__(256)
void gemm_wmma_kernel(const void* __restrict__ Av, const float* __restrict__ B,
                      float* __restrict__ C, int K, int ldb, int ldc)
{
    __shared__ unsigned short ldsB[64 * 32];   // [n][k] transposed, 4 KB

    const int lane = threadIdx.x & 31;
    const int w    = threadIdx.x >> 5;         // wave 0..7
    const int nb   = blockIdx.x * 64;
    const int ln15 = lane & 15;
    const int hi   = lane >> 4;                // 0/1
    const int kb8  = hi * 8;                   // A-frag k sub-base
    const int koff = hi * 16;                  // unused in this kernel
    const int mrow = blockIdx.y * 128 + w * 16 + ln15;

    v8f acc[4];
    #pragma unroll
    for (int t = 0; t < 4; ++t) acc[t] = zero8();

    const int nsteps = K >> 5;
    for (int kt = 0; kt < nsteps; ++kt) {
        // ---- stage B tile 32(K) x 64(N), transposed, f32 -> bf16 ----
        {
            const int e  = threadIdx.x * 8;
            const int k  = e >> 6;
            const int n0 = e & 63;
            const float* bp = B + (size_t)(kt * 32 + k) * ldb + nb + n0;
            float4 b0 = ((const float4*)bp)[0];
            float4 b1 = ((const float4*)bp)[1];
            ldsB[(n0 + 0) * 32 + k] = f2bf(b0.x);
            ldsB[(n0 + 1) * 32 + k] = f2bf(b0.y);
            ldsB[(n0 + 2) * 32 + k] = f2bf(b0.z);
            ldsB[(n0 + 3) * 32 + k] = f2bf(b0.w);
            ldsB[(n0 + 4) * 32 + k] = f2bf(b1.x);
            ldsB[(n0 + 5) * 32 + k] = f2bf(b1.y);
            ldsB[(n0 + 6) * 32 + k] = f2bf(b1.z);
            ldsB[(n0 + 7) * 32 + k] = f2bf(b1.w);
        }
        __syncthreads();

        // ---- A fragment (16x32 per wave) ----
        FragBF a;
        if (A_IS_BF16) {
            const unsigned short* ap =
                (const unsigned short*)Av + (size_t)mrow * K + kt * 32 + kb8;
            a.q[0] = ((const uint4*)ap)[0];          // K in [kb8, kb8+8)
            a.q[1] = ((const uint4*)(ap + 16))[0];   // K in [16+kb8, 16+kb8+8)
        } else {
            const float* ap = (const float*)Av + (size_t)mrow * K + kt * 32 + kb8;
            float4 x0 = ((const float4*)ap)[0];
            float4 x1 = ((const float4*)ap)[1];
            float4 y0 = ((const float4*)(ap + 16))[0];
            float4 y1 = ((const float4*)(ap + 16))[1];
            a.u[0] = pack2bf(x0.x, x0.y); a.u[1] = pack2bf(x0.z, x0.w);
            a.u[2] = pack2bf(x1.x, x1.y); a.u[3] = pack2bf(x1.z, x1.w);
            a.u[4] = pack2bf(y0.x, y0.y); a.u[5] = pack2bf(y0.z, y0.w);
            a.u[6] = pack2bf(y1.x, y1.y); a.u[7] = pack2bf(y1.z, y1.w);
        }

        // ---- 4 N-tiles of WMMA ----
        #pragma unroll
        for (int nt = 0; nt < 4; ++nt) {
            FragBF bf;
            const unsigned short* bp = &ldsB[(nt * 16 + ln15) * 32 + kb8];
            bf.q[0] = ((const uint4*)bp)[0];
            bf.q[1] = ((const uint4*)(bp + 16))[0];
            acc[nt] = wmma_bf16(a.v, bf.v, acc[nt]);
        }
        __syncthreads();
    }

    // ---- epilogue: C layout (VGPR r, lane l): M = r + 8*(l>>4), N = l&15 ----
    const int mtop = blockIdx.y * 128 + w * 16 + hi * 8;
    #pragma unroll
    for (int nt = 0; nt < 4; ++nt) {
        float* cp = C + (size_t)mtop * ldc + nb + nt * 16 + ln15;
        #pragma unroll
        for (int r = 0; r < 8; ++r)
            cp[(size_t)r * ldc] = acc[nt][r];
    }
}

// ---------------------------------------------------------------------------
// RoPE + bf16 conversion kernels
//   Qf f32 [b*2048+s][2048] -> Qb bf16 [b][h][s][128]      (rotated)
//   Kf f32 [b*2048+s][1024] -> Kb bf16 [b][kvh][s][128]    (rotated)
//   Vf f32 [b*2048+s][1024] -> Vt bf16 [b][kvh][128][s]    (transposed)
// ---------------------------------------------------------------------------
__global__ __launch_bounds__(256)
void rope_q_kernel(const float* __restrict__ Qf, const float* __restrict__ cosb,
                   const float* __restrict__ sinb, unsigned short* __restrict__ Qb)
{
    const int tid = blockIdx.x * 256 + threadIdx.x;      // 2*2048*16*64 threads
    const int i = tid & 63;
    const int h = (tid >> 6) & 15;
    const int s = (tid >> 10) & 2047;
    const int b = tid >> 21;
    const float* src = Qf + ((size_t)(b * 2048 + s)) * 2048 + h * 128 + 2 * i;
    const float xr = src[0], xi = src[1];
    const float c = cosb[s * 64 + i], sn = sinb[s * 64 + i];
    const size_t o = ((size_t)(b * 16 + h) * 2048 + s) * 128 + 2 * i;
    Qb[o]     = f2bf(xr * c - xi * sn);
    Qb[o + 1] = f2bf(xr * sn + xi * c);
}

__global__ __launch_bounds__(256)
void rope_k_kernel(const float* __restrict__ Kf, const float* __restrict__ cosb,
                   const float* __restrict__ sinb, unsigned short* __restrict__ Kb)
{
    const int tid = blockIdx.x * 256 + threadIdx.x;      // 2*2048*8*64 threads
    const int i  = tid & 63;
    const int kh = (tid >> 6) & 7;
    const int s  = (tid >> 9) & 2047;
    const int b  = tid >> 20;
    const float* src = Kf + ((size_t)(b * 2048 + s)) * 1024 + kh * 128 + 2 * i;
    const float xr = src[0], xi = src[1];
    const float c = cosb[s * 64 + i], sn = sinb[s * 64 + i];
    const size_t o = ((size_t)(b * 8 + kh) * 2048 + s) * 128 + 2 * i;
    Kb[o]     = f2bf(xr * c - xi * sn);
    Kb[o + 1] = f2bf(xr * sn + xi * c);
}

__global__ __launch_bounds__(256)
void conv_v_kernel(const float* __restrict__ Vf, unsigned short* __restrict__ Vt)
{
    const int tid = blockIdx.x * 256 + threadIdx.x;      // 2*8*128*2048 threads
    const int s  = tid & 2047;
    const int d  = (tid >> 11) & 127;
    const int kh = (tid >> 18) & 7;
    const int b  = tid >> 21;
    const float v = Vf[((size_t)(b * 2048 + s)) * 1024 + kh * 128 + d];
    Vt[((size_t)(b * 8 + kh) * 128 + d) * 2048 + s] = f2bf(v);
}

// ---------------------------------------------------------------------------
// Flash attention (causal, GQA n_rep=2). One wave per 16-row Q tile.
// K/V consumed in 32-column steps so P (16x32) is exactly one bf16 A-frag.
// Online softmax with 16-lane shfl_xor row reductions; P re-laid out C->A
// through a 1 KB per-wave LDS slice.
// ---------------------------------------------------------------------------
__global__ __launch_bounds__(128)
void flash_attn_kernel(const unsigned short* __restrict__ Qb,
                       const unsigned short* __restrict__ Kb,
                       const unsigned short* __restrict__ Vt,
                       unsigned short* __restrict__ AO)
{
    __shared__ unsigned short ldsP[4][16 * 32];          // per-wave P staging

    const int lane  = threadIdx.x & 31;
    const int w     = threadIdx.x >> 5;                  // wave 0..3
    const int b     = blockIdx.z;
    const int h     = blockIdx.y;
    const int kvh   = h >> 1;                            // GQA: 16 q-heads / 8 kv-heads
    const int qt    = blockIdx.x * 4 + w;
    const int qbase = qt * 16;
    const int ln15  = lane & 15;
    const int hi    = lane >> 4;
    const int kb8   = hi * 8;
    const int koff  = hi * 16;

    const unsigned short* qptr = Qb + ((size_t)(b * 16 + h) * 2048 + qbase) * 128;
    const unsigned short* kpt  = Kb + (size_t)(b * 8 + kvh) * 2048 * 128;
    const unsigned short* vpt  = Vt + (size_t)(b * 8 + kvh) * 128 * 2048;

    // Q fragments: 4 k-steps of 16x32 bf16
    FragBF qf[4];
    #pragma unroll
    for (int kk = 0; kk < 4; ++kk) {
        const unsigned short* p = qptr + (size_t)ln15 * 128 + kk * 32 + koff;
        qf[kk].q[0] = ((const uint4*)p)[0];
        qf[kk].q[1] = ((const uint4*)p)[1];
    }

    float rowm[8], rowl[8];
    v8f O[8];
    #pragma unroll
    for (int r = 0; r < 8; ++r) { rowm[r] = -1e30f; rowl[r] = 0.f; }
    #pragma unroll
    for (int nt = 0; nt < 8; ++nt) O[nt] = zero8();

    const float sc = 0.08838834764831845f;               // 1/sqrt(128)
    const int ntiles = (qbase + 47) >> 5;                // causal K coverage

    for (int j = 0; j < ntiles; ++j) {
        const int colbase = j * 32;

        // ---- S = Q K^T for 2 column tiles of 16 ----
        v8f S0 = zero8(), S1 = zero8();
        #pragma unroll
        for (int kk = 0; kk < 4; ++kk) {
            FragBF bk;
            const unsigned short* p0 =
                kpt + (size_t)(colbase + ln15) * 128 + kk * 32 + koff;
            bk.q[0] = ((const uint4*)p0)[0];
            bk.q[1] = ((const uint4*)p0)[1];
            S0 = wmma_bf16(qf[kk].v, bk.v, S0);
            const unsigned short* p1 = p0 + 16 * 128;
            bk.q[0] = ((const uint4*)p1)[0];
            bk.q[1] = ((const uint4*)p1)[1];
            S1 = wmma_bf16(qf[kk].v, bk.v, S1);
        }

        // ---- online softmax update (per C-layout row) ----
        #pragma unroll
        for (int r = 0; r < 8; ++r) {
            const int row = qbase + r + hi * 8;
            const int c0  = colbase + ln15;
            float s0 = (c0      <= row) ? S0[r] * sc : -1e30f;
            float s1 = (c0 + 16 <= row) ? S1[r] * sc : -1e30f;
            float mx = fmaxf(s0, s1);
            #pragma unroll
            for (int off = 1; off < 16; off <<= 1)
                mx = fmaxf(mx, __shfl_xor(mx, off, 32));
            const float mnew  = fmaxf(rowm[r], mx);
            const float alpha = __expf(rowm[r] - mnew);
            const float p0 = __expf(s0 - mnew);
            const float p1 = __expf(s1 - mnew);
            float sum = p0 + p1;
            #pragma unroll
            for (int off = 1; off < 16; off <<= 1)
                sum += __shfl_xor(sum, off, 32);
            rowl[r] = rowl[r] * alpha + sum;
            rowm[r] = mnew;
            #pragma unroll
            for (int nt = 0; nt < 8; ++nt) O[nt][r] *= alpha;
            const int m = r + hi * 8;
            ldsP[w][m * 32 + ln15]      = f2bf(p0);
            ldsP[w][m * 32 + 16 + ln15] = f2bf(p1);
        }
        __asm__ volatile("s_wait_dscnt 0" ::: "memory");

        // ---- P as 16x32 A-frag, then O += P * V ----
        FragBF pf;
        {
            const unsigned short* pp = &ldsP[w][ln15 * 32 + koff];
            pf.q[0] = ((const uint4*)pp)[0];
            pf.q[1] = ((const uint4*)pp)[1];
        }
        #pragma unroll
        for (int nt = 0; nt < 8; ++nt) {
            FragBF vf;
            const unsigned short* vp =
                vpt + (size_t)(nt * 16 + ln15) * 2048 + colbase + koff;
            vf.q[0] = ((const uint4*)vp)[0];
            vf.q[1] = ((const uint4*)vp)[1];
            O[nt] = wmma_bf16(pf.v, vf.v, O[nt]);
        }
    }

    // ---- finalize: divide by row sum, store bf16 [b][s][h*128+d] ----
    #pragma unroll
    for (int r = 0; r < 8; ++r) rowl[r] = 1.f / rowl[r];
    #pragma unroll
    for (int nt = 0; nt < 8; ++nt) {
        unsigned short* op = AO + ((size_t)b * 2048 + qbase + hi * 8) * 2048
                                + h * 128 + nt * 16 + ln15;
        #pragma unroll
        for (int r = 0; r < 8; ++r)
            op[(size_t)r * 2048] = f2bf(O[nt][r] * rowl[r]);
    }
}

// ---------------------------------------------------------------------------
// Launch
// ---------------------------------------------------------------------------
extern "C" void kernel_launch(void* const* d_in, const int* in_sizes, int n_in,
                              void* d_out, int out_size, void* d_ws, size_t ws_size,
                              hipStream_t stream)
{
    const float* hidden = (const float*)d_in[0];
    const float* fcos   = (const float*)d_in[1];
    const float* fsin   = (const float*)d_in[2];
    const float* wq     = (const float*)d_in[3];
    const float* wk     = (const float*)d_in[4];
    const float* wv     = (const float*)d_in[5];
    const float* wo     = (const float*)d_in[6];
    float* out = (float*)d_out;

    char* ws = (char*)d_ws;
    float*          Qf = (float*)(ws);                          // 32 MB
    float*          Kf = (float*)(ws + (size_t)(32u << 20));    // 16 MB
    float*          Vf = (float*)(ws + (size_t)(48u << 20));    // 16 MB
    unsigned short* Qb = (unsigned short*)(ws + (size_t)(64u << 20)); // 16 MB
    unsigned short* Kb = (unsigned short*)(ws + (size_t)(80u << 20)); //  8 MB
    unsigned short* Vt = (unsigned short*)(ws + (size_t)(88u << 20)); //  8 MB
    unsigned short* AO = (unsigned short*)(ws);                 // reuse Qf (16 MB)

    const dim3 blk(256);

    // QKV projections (f32 A -> bf16 WMMA, f32 accumulate)
    gemm_wmma_kernel<false><<<dim3(2048 / 64, 4096 / 128), blk, 0, stream>>>(
        hidden, wq, Qf, 2048, 2048, 2048);
    gemm_wmma_kernel<false><<<dim3(1024 / 64, 4096 / 128), blk, 0, stream>>>(
        hidden, wk, Kf, 2048, 1024, 1024);
    gemm_wmma_kernel<false><<<dim3(1024 / 64, 4096 / 128), blk, 0, stream>>>(
        hidden, wv, Vf, 2048, 1024, 1024);

    // RoPE + bf16 repacking (V transposed for contiguous B-frag loads)
    rope_q_kernel<<<16384, 256, 0, stream>>>(Qf, fcos, fsin, Qb);
    rope_k_kernel<<<8192, 256, 0, stream>>>(Kf, fcos, fsin, Kb);
    conv_v_kernel<<<16384, 256, 0, stream>>>(Vf, Vt);

    // Causal flash attention (wave per 16-row tile; 4 waves per block)
    flash_attn_kernel<<<dim3(32, 16, 2), dim3(128), 0, stream>>>(Qb, Kb, Vt, AO);

    // Output projection (bf16 A)
    gemm_wmma_kernel<true><<<dim3(2048 / 64, 4096 / 128), blk, 0, stream>>>(
        AO, wo, out, 2048, 2048, 2048);
}
